// LoraGatherBMM_59459527246490
// MI455X (gfx1250) — matrix operand order repaired and checked by reading, not compiled, not run on hardware
//
#include <hip/hip_runtime.h>

// Problem sizes (fixed by the reference)
#define B_DIM   2048
#define IN_DIM  4096
#define OUT_DIM 4096
#define RANK    16

// GEMM tiling
#define BM 128
#define BN 128
#define BK 32
#define NIT (IN_DIM / BK)   // 128 K-steps
#define LDP 40              // padded LDS row stride in halves (80B, multiple of 16B)
#define GDP 136             // G-tile row stride in halves (272B = 17*16B, 16B-aligned)

typedef _Float16 v16h __attribute__((ext_vector_type(16)));
typedef _Float16 v8h  __attribute__((ext_vector_type(8)));
typedef _Float16 v2h  __attribute__((ext_vector_type(2)));
typedef float    v8f  __attribute__((ext_vector_type(8)));
typedef int      v4i  __attribute__((ext_vector_type(4)));

#if defined(__has_builtin)
#if __has_builtin(__builtin_amdgcn_global_load_async_to_lds_b128)
#define USE_ASYNC_LDS 1
#endif
#endif

#ifdef USE_ASYNC_LDS
// Builtin signature (from hipcc diagnostic): (AS1 v4i*, AS3 v4i*, imm int, imm int)
#define TO_GLOBAL(p) ((__attribute__((address_space(1))) v4i*)(p))
#define TO_LDS(p)    ((__attribute__((address_space(3))) v4i*)(p))
#if __has_builtin(__builtin_amdgcn_s_wait_asynccnt)
#define WAIT_ASYNC(n) __builtin_amdgcn_s_wait_asynccnt(n)
#else
#define WAIT_ASYNC(n) asm volatile("s_wait_asynccnt %0" ::"i"(n) : "memory")
#endif
#else
#define WAIT_ASYNC(n)
#endif

// ---------------------------------------------------------------------------
// Kernel 1: h[b, r] = sum_i x[b,i] * lora_A[wid[b], i, r]   (f32 into d_ws)
// One wave per row b. 16 register accumulators, xor-shuffle reduction.
// ---------------------------------------------------------------------------
__global__ __launch_bounds__(256) void lora_h_kernel(
    const _Float16* __restrict__ x,
    const int*      __restrict__ wids,
    const _Float16* __restrict__ loraA,
    float*          __restrict__ H)
{
    const int wv   = threadIdx.x >> 5;
    const int lane = threadIdx.x & 31;
    const int b    = blockIdx.x * 8 + wv;

    const int wid = wids[b];
    const _Float16* xrow  = x + (size_t)b * IN_DIM;
    const _Float16* Abase = loraA + (size_t)wid * IN_DIM * RANK;

    float acc[RANK];
#pragma unroll
    for (int r = 0; r < RANK; ++r) acc[r] = 0.0f;

    for (int i = lane; i < IN_DIM; i += 32) {
        float xv = (float)xrow[i];
        const v8h* Ar = (const v8h*)(Abase + (size_t)i * RANK);
        v8h a0 = Ar[0];
        v8h a1 = Ar[1];
#pragma unroll
        for (int r = 0; r < 8; ++r) {
            acc[r]     += xv * (float)a0[r];
            acc[r + 8] += xv * (float)a1[r];
        }
    }

#pragma unroll
    for (int off = 16; off > 0; off >>= 1) {
#pragma unroll
        for (int r = 0; r < RANK; ++r)
            acc[r] += __shfl_xor(acc[r], off, 32);
    }

    if (lane == 0) {
#pragma unroll
        for (int r = 0; r < RANK; ++r)
            H[(size_t)b * RANK + r] = acc[r];
    }
}

// ---------------------------------------------------------------------------
// Kernel 2: out[b, o] = (x @ M)[b, o] + 2 * sum_r h[b,r] * lora_B[wid[b], r, o]
// 128x128 tile/block, 8 waves (each 32x64 = 2x4 WMMA accums), double-buffered
// LDS with async X staging, register-pipelined transposed M staging, and a
// vectorized LDS-staged LoRA epilogue.
// ---------------------------------------------------------------------------
__global__ __launch_bounds__(256) void lora_gemm_kernel(
    const _Float16* __restrict__ x,
    const int*      __restrict__ wids,
    const _Float16* __restrict__ loraB,
    const _Float16* __restrict__ Mw,
    const float*    __restrict__ H,
    _Float16*       __restrict__ out)
{
    __shared__ __align__(16) _Float16 Xs[2][BM][LDP];  // A tile, row-major [m][k]
    __shared__ __align__(16) _Float16 Ms[2][BN][LDP];  // B tile, transposed [n][k]
    __shared__ __align__(16) _Float16 Gs[BM][GDP];     // 2*(h @ lora_B) tile

    const int t     = threadIdx.x;
    const int lane  = t & 31;
    const int wv    = t >> 5;       // 0..7
    const int waveM = wv & 3;       // 0..3  -> 32 rows each
    const int waveN = wv >> 2;      // 0..1  -> 64 cols each
    const int lhalf = lane >> 4;    // 0 or 1
    const int lmod  = lane & 15;

    const int rowBase = blockIdx.y * BM;
    const int colBase = blockIdx.x * BN;

    // Per-thread staging coordinates.
    // X tile (128x32): 2 chunks of 8 halves: rows xr0, xr0+64, cols xc0..xc0+7
    const int xr0 = t >> 2;
    const int xc0 = (t & 3) * 8;
    // M tile (32x128): 2 K-rows x 8 N-cols per thread, packed b32 stores
    const int kr = (t >> 4) * 2;    // 0,2,..,30
    const int nc = (t & 15) * 8;    // 0..120

    const _Float16* xBase = x + (size_t)rowBase * IN_DIM;

    v8f C[2][4];
#pragma unroll
    for (int mi = 0; mi < 2; ++mi)
#pragma unroll
        for (int ni = 0; ni < 4; ++ni)
            C[mi][ni] = (v8f){};

    // ---- prologue: stage tile 0 into buffer 0 ----
#pragma unroll
    for (int s = 0; s < 2; ++s) {
        const _Float16* src = xBase + (size_t)(xr0 + s * 64) * IN_DIM + xc0;
#ifdef USE_ASYNC_LDS
        __builtin_amdgcn_global_load_async_to_lds_b128(
            TO_GLOBAL(src), TO_LDS(&Xs[0][xr0 + s * 64][xc0]), 0, 0);
#else
        *(v8h*)&Xs[0][xr0 + s * 64][xc0] = *(const v8h*)src;
#endif
    }
    {
        const _Float16* msrc = Mw + (size_t)kr * OUT_DIM + colBase + nc;
        v8h a = *(const v8h*)msrc;
        v8h b = *(const v8h*)(msrc + OUT_DIM);
#pragma unroll
        for (int j = 0; j < 8; ++j) {
            v2h p = {a[j], b[j]};
            *(v2h*)&Ms[0][nc + j][kr] = p;
        }
    }

    v8h mra = {}, mrb = {};
#ifndef USE_ASYNC_LDS
    v8h xrg[2];
#endif

    for (int i = 0; i < NIT; ++i) {
        const int  cur     = i & 1;
        const int  nxt     = cur ^ 1;
        const bool hasNext = (i + 1) < NIT;
        const int  kN      = (i + 1) * BK;

        // ---- stage tile i+1 (X async -> LDS, M global -> regs) ----
        if (hasNext) {
            const int kPF = (kN + BK) & (IN_DIM - 1);   // warm tile i+2 in L2
#pragma unroll
            for (int s = 0; s < 2; ++s) {
                const _Float16* src = xBase + (size_t)(xr0 + s * 64) * IN_DIM + kN + xc0;
                __builtin_prefetch(xBase + (size_t)(xr0 + s * 64) * IN_DIM + kPF + xc0, 0, 1);
#ifdef USE_ASYNC_LDS
                __builtin_amdgcn_global_load_async_to_lds_b128(
                    TO_GLOBAL(src), TO_LDS(&Xs[nxt][xr0 + s * 64][xc0]), 0, 0);
#else
                xrg[s] = *(const v8h*)src;
#endif
            }
            const _Float16* msrc = Mw + (size_t)(kN + kr) * OUT_DIM + colBase + nc;
            __builtin_prefetch(Mw + (size_t)(kPF + kr) * OUT_DIM + colBase + nc, 0, 1);
            mra = *(const v8h*)msrc;
            mrb = *(const v8h*)(msrc + OUT_DIM);
        }

        // retire tile i's async loads (leave tile i+1's 2 in flight)
        if (hasNext) { WAIT_ASYNC(2); } else { WAIT_ASYNC(0); }
        __syncthreads();

        // ---- fragments from buffer `cur` ----
        // A (16x32 f16): lane<16 holds K {0..7,16..23}, lane>=16 holds {8..15,24..31}
        v16h Af[2];
#pragma unroll
        for (int mi = 0; mi < 2; ++mi) {
            int m = waveM * 32 + mi * 16 + lmod;
            v8h lo = *(const v8h*)&Xs[cur][m][lhalf * 8];
            v8h hi = *(const v8h*)&Xs[cur][m][16 + lhalf * 8];
            Af[mi] = __builtin_shufflevector(lo, hi,
                0, 1, 2, 3, 4, 5, 6, 7, 8, 9, 10, 11, 12, 13, 14, 15);
        }
        // B (32x16 f16): lane n = lmod; lane<16 holds K {0..7,16..23}, lane>=16 holds {8..15,24..31}
        v16h Bf[4];
#pragma unroll
        for (int ni = 0; ni < 4; ++ni) {
            int n = waveN * 64 + ni * 16 + lmod;
            v8h lo = *(const v8h*)&Ms[cur][n][lhalf * 8];
            v8h hi = *(const v8h*)&Ms[cur][n][16 + lhalf * 8];
            Bf[ni] = __builtin_shufflevector(lo, hi,
                0, 1, 2, 3, 4, 5, 6, 7, 8, 9, 10, 11, 12, 13, 14, 15);
        }

        // ---- 8 WMMAs ----
#pragma unroll
        for (int mi = 0; mi < 2; ++mi)
#pragma unroll
            for (int ni = 0; ni < 4; ++ni)
                C[mi][ni] = __builtin_amdgcn_wmma_f32_16x16x32_f16(
                    false, Af[mi], false, Bf[ni],
                    (short)0, C[mi][ni], false, false);

        // ---- drain staged M (and X fallback) regs into buffer `nxt` ----
        if (hasNext) {
#ifndef USE_ASYNC_LDS
#pragma unroll
            for (int s = 0; s < 2; ++s)
                *(v8h*)&Xs[nxt][xr0 + s * 64][xc0] = xrg[s];
#endif
#pragma unroll
            for (int j = 0; j < 8; ++j) {
                v2h p = {mra[j], mrb[j]};
                *(v2h*)&Ms[nxt][nc + j][kr] = p;
            }
        }
        __syncthreads();
    }

    // ---- epilogue stage 1: Gs = 2 * (h @ lora_B[wid]) tile, vectorized ----
    {
        const int ncG = (t & 15) * 8;   // 8 consecutive cols
        const int rG0 = (t >> 4) * 8;   // 8 consecutive rows
#pragma unroll 1
        for (int rr = 0; rr < 8; ++rr) {
            int m   = rG0 + rr;
            int gm  = rowBase + m;
            int wid = wids[gm];
            const float*    Hp = H + (size_t)gm * RANK;
            const _Float16* Bp = loraB + (size_t)wid * RANK * OUT_DIM + colBase + ncG;
            float acc[8];
#pragma unroll
            for (int j = 0; j < 8; ++j) acc[j] = 0.0f;
#pragma unroll
            for (int r = 0; r < RANK; ++r) {
                v8h bv = *(const v8h*)(Bp + (size_t)r * OUT_DIM);
                float hr = Hp[r];
#pragma unroll
                for (int j = 0; j < 8; ++j) acc[j] += hr * (float)bv[j];
            }
            v8h g;
#pragma unroll
            for (int j = 0; j < 8; ++j) g[j] = (_Float16)(2.0f * acc[j]);
            *(v8h*)&Gs[m][ncG] = g;
        }
    }
    __syncthreads();

    // ---- epilogue stage 2: out = C + Gs ----
#pragma unroll
    for (int mi = 0; mi < 2; ++mi) {
#pragma unroll
        for (int ni = 0; ni < 4; ++ni) {
            int nl = waveN * 64 + ni * 16 + lmod;
            int gn = colBase + nl;
#pragma unroll
            for (int v = 0; v < 8; ++v) {
                int ml = waveM * 32 + mi * 16 + lhalf * 8 + v;
                int gm = rowBase + ml;
                out[(size_t)gm * OUT_DIM + gn] =
                    (_Float16)(C[mi][ni][v] + (float)Gs[ml][nl]);
            }
        }
    }
}

// ---------------------------------------------------------------------------
extern "C" void kernel_launch(void* const* d_in, const int* in_sizes, int n_in,
                              void* d_out, int out_size, void* d_ws, size_t ws_size,
                              hipStream_t stream) {
    (void)in_sizes; (void)n_in; (void)out_size; (void)ws_size;
    const _Float16* x     = (const _Float16*)d_in[0];
    const int*      wids  = (const int*)d_in[1];
    const _Float16* loraA = (const _Float16*)d_in[2];
    const _Float16* loraB = (const _Float16*)d_in[3];
    const _Float16* M     = (const _Float16*)d_in[4];
    _Float16*       out   = (_Float16*)d_out;
    float*          H     = (float*)d_ws;   // 2048*16 f32 = 128 KB

    lora_h_kernel<<<dim3(B_DIM / 8), dim3(256), 0, stream>>>(x, wids, loraA, H);
    lora_gemm_kernel<<<dim3(OUT_DIM / BN, B_DIM / BM), dim3(256), 0, stream>>>(
        x, wids, loraB, M, H, out);
}
